// data2vec_multo_model_export_56684978373082
// MI455X (gfx1250) — matrix operand order repaired and checked by reading, not compiled, not run on hardware
//
#include <hip/hip_runtime.h>
#include <hip/hip_bf16.h>

#define NB   2
#define TRAW 8000
#define NF   40
#define C1C  512
#define ND   768
#define NT1  3999
#define NT   1999
#define NH   12
#define DHD  64
#define NL   8
#define NFF  3072
#define NV   8000
#define BTROWS (NB*NT)

typedef __attribute__((ext_vector_type(16))) __bf16 v16bf;
typedef __attribute__((ext_vector_type(8)))  float  v8f;
typedef __attribute__((ext_vector_type(4)))  unsigned v4u;
typedef __attribute__((ext_vector_type(8)))  int    v8i;
typedef __attribute__((ext_vector_type(4)))  int    v4i;

__device__ __forceinline__ unsigned short f2bfu(float f) {
  return __builtin_bit_cast(unsigned short, (__bf16)f);   // RNE
}
__device__ __forceinline__ __bf16 us2bf(unsigned short u) {
  return __builtin_bit_cast(__bf16, u);
}
__device__ __forceinline__ float gelu_tanh(float x) {
  float x3 = x * x * x;
  return 0.5f * x * (1.0f + tanhf(0.7978845608028654f * (x + 0.044715f * x3)));
}
__device__ __forceinline__ v8f wmma_bf16(v16bf a, v16bf b, v8f c) {
  return __builtin_amdgcn_wmma_f32_16x16x32_bf16(false, a, false, b, (short)0, c, false, false);
}
__device__ __forceinline__ unsigned lds_off(const void* p) {
  return (unsigned)(unsigned long long)p;   // low 32 bits = LDS byte address
}
__device__ __forceinline__ void async_b128(unsigned ldsaddr, const void* gaddr) {
  asm volatile("global_load_async_to_lds_b128 %0, %1, off"
               :: "v"(ldsaddr), "v"(gaddr) : "memory");
}
__device__ __forceinline__ void wait_async0() {
  asm volatile("s_wait_asynccnt 0x0" ::: "memory");
}

// Tensor Data Mover: DMA a 2D tile [rows x rowlen] of bf16 (2B elems) from a
// row-major tensor (row stride = stride elems) into LDS, with LDS padding of
// 4 DWORDs after every 32 DWORDs (row of 64 elems) -> LDS row stride 72 ushorts.
__device__ __forceinline__ void tdm_load_2d(unsigned ldsaddr, const void* gaddr,
                                            int rows, int rowlen,
                                            int tensor_rows, int tensor_cols,
                                            int stride) {
  unsigned long long ga = (unsigned long long)gaddr;
  v4u g0;
  g0[0] = 1u;                                              // count=1 (valid), user mode
  g0[1] = ldsaddr;                                         // lds_addr
  g0[2] = (unsigned)(ga & 0xffffffffu);                    // global_addr[31:0]
  g0[3] = (unsigned)((ga >> 32) & 0x01ffffffu) | (2u << 30); // global_addr[56:32], type=2
  v8i g1;
  g1[0] = (int)((1u << 16) | (1u << 20) | (4u << 22) | (3u << 25));
  // data_size=1(2B) | pad_enable | pad_interval=32dw | pad_amount=4dw
  g1[1] = (int)(((unsigned)tensor_cols & 0xffffu) << 16);  // tensor_dim0[15:0]
  g1[2] = (int)((((unsigned)tensor_cols >> 16) & 0xffffu) |
                (((unsigned)tensor_rows & 0xffffu) << 16)); // tensor_dim0[31:16], tensor_dim1[15:0]
  g1[3] = (int)((((unsigned)tensor_rows >> 16) & 0xffffu) |
                (((unsigned)rowlen & 0xffffu) << 16));      // tensor_dim1[31:16], tile_dim0
  g1[4] = (int)((unsigned)rows & 0xffffu);                  // tile_dim1 (tile_dim2=0)
  g1[5] = stride;                                           // tensor_dim0_stride[31:0]
  g1[6] = 0;
  g1[7] = 0;
  v4i z4 = {0, 0, 0, 0};
  v8i z8 = {0, 0, 0, 0, 0, 0, 0, 0};
  __builtin_amdgcn_tensor_load_to_lds(g0, g1, z4, z4, z8, 0);
}

// ---------------------------------------------------------------- fp32->bf16
__global__ void k_f32_to_bf16(const float* __restrict__ s,
                              unsigned short* __restrict__ d, long n) {
  long i = (long)blockIdx.x * blockDim.x + threadIdx.x;
  long st = (long)gridDim.x * blockDim.x;
  for (; i < n; i += st) d[i] = f2bfu(s[i]);
}

// ---------------------------------------------------------------- conv1 (40->512, k=3, s=2) + gelu
__global__ void k_conv1(const float* __restrict__ feats,   // [NB][TRAW][NF]
                        const float* __restrict__ w,       // [C1C][NF][3]
                        const float* __restrict__ bias,
                        float* __restrict__ out) {         // [NB][NT1][C1C]
  __shared__ float sf[3 * NF];
  int bt = blockIdx.x;
  int b = bt / NT1, t1 = bt % NT1;
  const float* fr = feats + ((long)b * TRAW + 2 * t1) * NF;   // 3 rows contiguous
  for (int i = threadIdx.x; i < 3 * NF; i += blockDim.x) sf[i] = fr[i];
  __syncthreads();
  for (int c = threadIdx.x; c < C1C; c += blockDim.x) {
    const float* wc = w + (long)c * NF * 3;
    float s = bias[c];
    #pragma unroll
    for (int k = 0; k < 3; k++)
      for (int f = 0; f < NF; f++)
        s += sf[k * NF + f] * wc[f * 3 + k];
    out[((long)b * NT1 + t1) * C1C + c] = gelu_tanh(s);
  }
}

// ---------------------------------------------------------------- conv2 (512->768, k=3, s=2) + gelu
__global__ void k_conv2(const float* __restrict__ c1,      // [NB][NT1][C1C]
                        const float* __restrict__ w,       // [ND][C1C][3]
                        const float* __restrict__ bias,
                        float* __restrict__ out) {         // [NB][NT][ND]
  __shared__ float sc[3 * C1C];
  int bt = blockIdx.x;
  int b = bt / NT, t = bt % NT;
  const float* cr = c1 + ((long)b * NT1 + 2 * t) * C1C;    // 3 rows contiguous
  for (int i = threadIdx.x; i < 3 * C1C; i += blockDim.x) sc[i] = cr[i];
  __syncthreads();
  for (int d = threadIdx.x; d < ND; d += blockDim.x) {
    const float* wd = w + (long)d * C1C * 3;
    float s = bias[d];
    for (int c = 0; c < C1C; c++) {
      s += sc[c] * wd[c * 3 + 0] + sc[C1C + c] * wd[c * 3 + 1] + sc[2 * C1C + c] * wd[c * 3 + 2];
    }
    out[((long)b * NT + t) * ND + d] = gelu_tanh(s);
  }
}

// ---------------------------------------------------------------- layernorm (row = 768), f32 or bf16 out
__global__ void k_ln(const float* __restrict__ in, const float* __restrict__ g,
                     const float* __restrict__ bb, void* __restrict__ out, int outBf16) {
  __shared__ float red[256];
  int row = blockIdx.x;
  const float* xr = in + (long)row * ND;
  int tid = threadIdx.x;
  float s = 0.f;
  for (int i = tid; i < ND; i += 256) s += xr[i];
  red[tid] = s; __syncthreads();
  for (int o = 128; o > 0; o >>= 1) { if (tid < o) red[tid] += red[tid + o]; __syncthreads(); }
  float mu = red[0] * (1.0f / ND);
  __syncthreads();
  float v = 0.f;
  for (int i = tid; i < ND; i += 256) { float d0 = xr[i] - mu; v += d0 * d0; }
  red[tid] = v; __syncthreads();
  for (int o = 128; o > 0; o >>= 1) { if (tid < o) red[tid] += red[tid + o]; __syncthreads(); }
  float inv = rsqrtf(red[0] * (1.0f / ND) + 1e-5f);
  for (int i = tid; i < ND; i += 256) {
    float y = (xr[i] - mu) * inv * g[i] + bb[i];
    if (outBf16) ((unsigned short*)out)[(long)row * ND + i] = f2bfu(y);
    else         ((float*)out)[(long)row * ND + i] = y;
  }
}

// ---------------------------------------------------------------- grouped pos conv (K=95, SAME) + gelu + residual
__global__ void k_posconv(const float* __restrict__ x,     // [NB][NT][ND]
                          const float* __restrict__ w,     // [ND][48][95]
                          const float* __restrict__ bias,
                          float* __restrict__ out) {
  long total = (long)BTROWS * ND;
  long idx = (long)blockIdx.x * blockDim.x + threadIdx.x;
  long st = (long)gridDim.x * blockDim.x;
  for (; idx < total; idx += st) {
    int d = (int)(idx % ND);
    long bt = idx / ND;
    int t = (int)(bt % NT), b = (int)(bt / NT);
    int g = d / 48;
    const float* wd = w + (long)d * 48 * 95;
    const float* xb = x + ((long)b * NT) * ND + g * 48;
    float s = bias[d];
    int k0 = (47 - t > 0) ? (47 - t) : 0;
    int k1 = (NT + 47 - t < 95) ? (NT + 47 - t) : 95;
    for (int k = k0; k < k1; k++) {
      const float* xr = xb + (long)(t + k - 47) * ND;
      for (int ci = 0; ci < 48; ci++) s += xr[ci] * wd[ci * 95 + k];
    }
    out[idx] = x[idx] + gelu_tanh(s);
  }
}

// ---------------------------------------------------------------- WMMA GEMM: C = act(A @ W^T + bias) [+res]
// A: bf16 [M][K]; W: bf16 [N][K]; C: f32 or bf16 [M][N]; optional row perm b*T+t -> t*B+b
// Double-buffered TDM pipeline: wave0 DMAs W tiles, wave1 DMAs A tiles; the
// DMA for tile t+1 is issued right after the barrier that publishes tile t,
// overlapping TDM with the WMMA stream. K is a multiple of BK in all calls.
#define BM 128
#define BN 128
#define BK 64
__global__ __launch_bounds__(256)
void k_gemm(const unsigned short* __restrict__ A, const unsigned short* __restrict__ W,
            const float* __restrict__ bias, const float* __restrict__ res,
            void* __restrict__ Cout, int M, int N, int K,
            int act, int outBf16, int permTB) {
  __shared__ unsigned short As[2][BM][BK + 8];   // stride 144B (TDM pad 4dw/32dw)
  __shared__ unsigned short Bs[2][BN][BK + 8];
  int tid = threadIdx.x;
  int lane = tid & 31, wid = tid >> 5;
  int half = lane >> 4, l16 = lane & 15;
  int bm = blockIdx.y * BM, bn = blockIdx.x * BN;
  int waveM = (wid >> 1) * 32, waveN = (wid & 1) * 64;
  int widu = __builtin_amdgcn_readfirstlane(wid);   // scalar branch for TDM issue

  v8f acc[2][4];
  #pragma unroll
  for (int i = 0; i < 2; i++)
    #pragma unroll
    for (int j = 0; j < 4; j++) acc[i][j] = (v8f){0,0,0,0,0,0,0,0};

  unsigned lA[2] = { lds_off(&As[0][0][0]), lds_off(&As[1][0][0]) };
  unsigned lB[2] = { lds_off(&Bs[0][0][0]), lds_off(&Bs[1][0][0]) };
  const unsigned short* Wp = W + (long)bn * K;
  const unsigned short* Ap = A + (long)bm * K;
  int nt = K / BK;

  // preload tile 0
  if (widu == 0)      tdm_load_2d(lB[0], Wp, BN, BK, N - bn, K, K);
  else if (widu == 1) tdm_load_2d(lA[0], Ap, BM, BK, M - bm, K, K);

  for (int t = 0; t < nt; ++t) {
    int cur = t & 1;
    if (widu < 2) __builtin_amdgcn_s_wait_tensorcnt(0);   // tile t DMA complete
    __syncthreads();                                       // publish tile t; all readers of buf[1-cur] done
    if (t + 1 < nt) {                                      // DMA tile t+1 overlaps compute of tile t
      if (widu == 0)      tdm_load_2d(lB[1 - cur], Wp + (long)(t + 1) * BK, BN, BK, N - bn, K, K);
      else if (widu == 1) tdm_load_2d(lA[1 - cur], Ap + (long)(t + 1) * BK, BM, BK, M - bm, K, K);
    }

    #pragma unroll
    for (int c = 0; c < 2; c++) {     // two 32-wide K chunks
      v16bf afr[2], bfr[4];
      #pragma unroll
      for (int i = 0; i < 2; i++) {
        const unsigned short* p = &As[cur][waveM + i * 16 + l16][c * 32];
        #pragma unroll
        for (int e = 0; e < 8; e++) afr[i][e] = us2bf(p[half * 8 + e]);
        #pragma unroll
        for (int e = 0; e < 8; e++) afr[i][8 + e] = us2bf(p[16 + half * 8 + e]);
      }
      #pragma unroll
      for (int j = 0; j < 4; j++) {
        const unsigned short* p = &Bs[cur][waveN + j * 16 + l16][c * 32];
        #pragma unroll
        for (int e = 0; e < 16; e++) bfr[j][e] = us2bf(p[(e < 8 ? 0 : 8) + half * 8 + e]);
      }
      #pragma unroll
      for (int i = 0; i < 2; i++)
        #pragma unroll
        for (int j = 0; j < 4; j++)
          acc[i][j] = wmma_bf16(afr[i], bfr[j], acc[i][j]);
    }
  }

  // epilogue: C-layout row = r + 8*half, col = l16
  #pragma unroll
  for (int i = 0; i < 2; i++) {
    #pragma unroll
    for (int j = 0; j < 4; j++) {
      int col = bn + waveN + j * 16 + l16;
      #pragma unroll
      for (int r = 0; r < 8; r++) {
        int row = bm + waveM + i * 16 + r + 8 * half;
        if (row < M && col < N) {
          float v = acc[i][j][r] + bias[col];
          if (act == 1) v = gelu_tanh(v);
          if (res) v += res[(long)row * N + col];
          long orow = permTB ? (long)(row % NT) * NB + (row / NT) : (long)row;
          if (outBf16) ((unsigned short*)Cout)[orow * N + col] = f2bfu(v);
          else         ((float*)Cout)[orow * N + col] = v;
        }
      }
    }
  }
}

// ---------------------------------------------------------------- flash attention with ALiBi (WMMA bf16)
// qkv: bf16 [NB][NT][3*ND] (q|k|v), o: bf16 [NB][NT][ND]. block=128 (4 waves, 16 q-rows each)
__global__ __launch_bounds__(128)
void k_attn(const unsigned short* __restrict__ qkv, unsigned short* __restrict__ o,
            const float* __restrict__ alibi_scale) {
  __shared__ unsigned short Ks[32][72];        // [k-rel][dh]
  __shared__ unsigned short Vts[64][40];       // [dh][k-rel] (transposed)
  __shared__ unsigned short Plds[4][16][32];   // per-wave P tile

  int tid = threadIdx.x;
  int wid = tid >> 5, lane = tid & 31;
  int half = lane >> 4, l16 = lane & 15;
  int b = blockIdx.z, h = blockIdx.y;
  int qb = blockIdx.x * 64 + wid * 16;
  const int ld = 3 * ND;
  const unsigned short* qbase = qkv + (long)b * NT * ld + h * DHD;
  const unsigned short* kbase = qbase + ND;
  const unsigned short* vbase = qbase + 2 * ND;
  float asc = alibi_scale[h]; if (asc < 0.f) asc = 0.f;
  float slope = exp2f(-8.0f * (float)(h + 1) / (float)NH) * asc;
  const float sm = 0.125f; // 1/sqrt(64)

  // Q fragments (held for the whole loop)
  int qrow = qb + l16; if (qrow > NT - 1) qrow = NT - 1;
  const unsigned short* qp = qbase + (long)qrow * ld;
  v16bf aq[2];
  #pragma unroll
  for (int c = 0; c < 2; c++) {
    #pragma unroll
    for (int e = 0; e < 8; e++) aq[c][e] = us2bf(qp[c * 32 + half * 8 + e]);
    #pragma unroll
    for (int e = 0; e < 8; e++) aq[c][8 + e] = us2bf(qp[c * 32 + 16 + half * 8 + e]);
  }

  v8f oacc[4];
  #pragma unroll
  for (int t = 0; t < 4; t++) oacc[t] = (v8f){0,0,0,0,0,0,0,0};
  float rmax[8], rsum[8];
  #pragma unroll
  for (int r = 0; r < 8; r++) { rmax[r] = -1e30f; rsum[r] = 0.f; }

  int kk = tid >> 2;            // 0..31
  int seg = (tid & 3) * 16;     // 0/16/32/48
  unsigned lK = lds_off(&Ks[kk][seg]);

  for (int kt = 0; kt < NT; kt += 32) {
    __syncthreads();            // prior iteration done with Ks/Vts
    int krow = kt + kk; if (krow > NT - 1) krow = NT - 1;
    const unsigned short* kp = kbase + (long)krow * ld + seg;
    const unsigned short* vp = vbase + (long)krow * ld + seg;
    async_b128(lK, kp);                       // K tile: async direct-to-LDS
    async_b128(lK + 16, kp + 8);
    #pragma unroll
    for (int e = 0; e < 16; e++) Vts[seg + e][kk] = vp[e];   // V transposed (scatter)
    wait_async0();
    __syncthreads();

    // S = Q K^T (two 16-col subtiles, dh split into 2x32)
    v8f S[2];
    S[0] = (v8f){0,0,0,0,0,0,0,0};
    S[1] = (v8f){0,0,0,0,0,0,0,0};
    #pragma unroll
    for (int j = 0; j < 2; j++) {
      #pragma unroll
      for (int c = 0; c < 2; c++) {
        const unsigned short* p = &Ks[j * 16 + l16][c * 32 + half * 8];
        v16bf bk;
        #pragma unroll
        for (int e = 0; e < 16; e++) bk[e] = us2bf(p[(e < 8 ? 0 : 8) + e]);
        S[j] = wmma_bf16(aq[c], bk, S[j]);
      }
    }

    // scale + ALiBi + column mask
    int kcol[2] = { kt + l16, kt + 16 + l16 };
    #pragma unroll
    for (int j = 0; j < 2; j++) {
      bool ok = kcol[j] < NT;
      #pragma unroll
      for (int r = 0; r < 8; r++) {
        int row = qb + r + 8 * half;
        float s = S[j][r] * sm - slope * fabsf((float)(row - kcol[j]));
        S[j][r] = ok ? s : -1e30f;
      }
    }

    // online softmax over this 32-col tile
    float corr[8];
    #pragma unroll
    for (int r = 0; r < 8; r++) {
      float m = fmaxf(S[0][r], S[1][r]);
      #pragma unroll
      for (int msk = 1; msk <= 8; msk <<= 1) m = fmaxf(m, __shfl_xor(m, msk, 32));
      float mn = fmaxf(rmax[r], m);
      corr[r] = __expf(rmax[r] - mn);
      rmax[r] = mn;
    }
    #pragma unroll
    for (int j = 0; j < 2; j++)
      #pragma unroll
      for (int r = 0; r < 8; r++)
        S[j][r] = __expf(S[j][r] - rmax[r]);
    #pragma unroll
    for (int r = 0; r < 8; r++) {
      float s = S[0][r] + S[1][r];
      #pragma unroll
      for (int msk = 1; msk <= 8; msk <<= 1) s += __shfl_xor(s, msk, 32);
      rsum[r] = rsum[r] * corr[r] + s;
    }
    #pragma unroll
    for (int t = 0; t < 4; t++)
      #pragma unroll
      for (int r = 0; r < 8; r++) oacc[t][r] *= corr[r];

    // P: C-layout -> LDS -> A-layout (wave-local, in-order LDS)
    #pragma unroll
    for (int j = 0; j < 2; j++)
      #pragma unroll
      for (int r = 0; r < 8; r++)
        Plds[wid][r + 8 * half][j * 16 + l16] = f2bfu(S[j][r]);
    v16bf ap;
    {
      const unsigned short* p = &Plds[wid][l16][0];
      #pragma unroll
      for (int e = 0; e < 8; e++) ap[e] = us2bf(p[half * 8 + e]);
      #pragma unroll
      for (int e = 0; e < 8; e++) ap[8 + e] = us2bf(p[16 + half * 8 + e]);
    }

    // O += P @ V  (4 dh tiles)
    #pragma unroll
    for (int t = 0; t < 4; t++) {
      const unsigned short* p = &Vts[t * 16 + l16][half * 8];
      v16bf bv;
      #pragma unroll
      for (int e = 0; e < 16; e++) bv[e] = us2bf(p[(e < 8 ? 0 : 8) + e]);
      oacc[t] = wmma_bf16(ap, bv, oacc[t]);
    }
  }

  // write O / rowsum (bf16)
  #pragma unroll
  for (int t = 0; t < 4; t++) {
    #pragma unroll
    for (int r = 0; r < 8; r++) {
      int row = qb + r + 8 * half;
      if (row < NT)
        o[((long)b * NT + row) * ND + h * DHD + t * 16 + l16] = f2bfu(oacc[t][r] / rsum[r]);
    }
  }
}

// ================================================================ host
extern "C" void kernel_launch(void* const* d_in, const int* in_sizes, int n_in,
                              void* d_out, int out_size, void* d_ws, size_t ws_size,
                              hipStream_t stream) {
  (void)in_sizes; (void)n_in; (void)out_size; (void)ws_size;
  const float* feats   = (const float*)d_in[0];
  const float* conv1_w = (const float*)d_in[1];
  const float* conv1_b = (const float*)d_in[2];
  const float* conv2_w = (const float*)d_in[3];
  const float* conv2_b = (const float*)d_in[4];
  const float* feat_g  = (const float*)d_in[5];
  const float* feat_b  = (const float*)d_in[6];
  const float* pos_w   = (const float*)d_in[7];
  const float* pos_b   = (const float*)d_in[8];
  const float* ctx_g   = (const float*)d_in[9];
  const float* ctx_b   = (const float*)d_in[10];
  const float* alibi   = (const float*)d_in[11];
  const float* Wqkv    = (const float*)d_in[12];
  const float* bqkv    = (const float*)d_in[13];
  const float* Wo      = (const float*)d_in[14];
  const float* bo      = (const float*)d_in[15];
  const float* ln1g    = (const float*)d_in[16];
  const float* ln1b    = (const float*)d_in[17];
  const float* ln2g    = (const float*)d_in[18];
  const float* ln2b    = (const float*)d_in[19];
  const float* W1      = (const float*)d_in[20];
  const float* b1      = (const float*)d_in[21];
  const float* W2      = (const float*)d_in[22];
  const float* b2      = (const float*)d_in[23];
  const float* projw   = (const float*)d_in[24];
  const float* projb   = (const float*)d_in[25];

  char* ws = (char*)d_ws;
  size_t off = 0;
  auto alloc = [&](size_t bytes) -> void* {
    void* p = ws + off;
    off = (off + bytes + 255) & ~(size_t)255;
    return p;
  };
  float* bufX = (float*)alloc((size_t)BTROWS * ND * 4);           // residual stream (f32)
  float* bufA = (float*)alloc((size_t)BTROWS * ND * 4);           // conv/posconv scratch (f32)
  unsigned short* bufH   = (unsigned short*)alloc((size_t)BTROWS * ND * 2);      // LN out (bf16)
  unsigned short* bufQKV = (unsigned short*)alloc((size_t)BTROWS * 3 * ND * 2);  // bf16
  unsigned short* bufO   = (unsigned short*)alloc((size_t)BTROWS * ND * 2);      // bf16
  unsigned short* bufFF  = (unsigned short*)alloc((size_t)BTROWS * NFF * 2);     // bf16
  float* c1buf = (float*)alloc((size_t)NB * NT1 * C1C * 4);
  unsigned short* wqkv_bf = (unsigned short*)alloc((size_t)NL * 3 * ND * ND * 2);
  unsigned short* wo_bf   = (unsigned short*)alloc((size_t)NL * ND * ND * 2);
  unsigned short* w1_bf   = (unsigned short*)alloc((size_t)NL * NFF * ND * 2);
  unsigned short* w2_bf   = (unsigned short*)alloc((size_t)NL * ND * NFF * 2);
  unsigned short* proj_bf = (unsigned short*)alloc((size_t)NV * ND * 2);

  auto conv = [&](const float* s, unsigned short* d, long n) {
    k_f32_to_bf16<<<2048, 256, 0, stream>>>(s, d, n);
  };
  conv(Wqkv,  wqkv_bf, (long)NL * 3 * ND * ND);
  conv(Wo,    wo_bf,   (long)NL * ND * ND);
  conv(W1,    w1_bf,   (long)NL * NFF * ND);
  conv(W2,    w2_bf,   (long)NL * ND * NFF);
  conv(projw, proj_bf, (long)NV * ND);

  auto gemm = [&](const unsigned short* A, const unsigned short* Wb, const float* bias,
                  const float* res, void* Cp, int M, int N, int K,
                  int act, int outBf16, int permTB) {
    dim3 g((N + BN - 1) / BN, (M + BM - 1) / BM);
    k_gemm<<<g, 256, 0, stream>>>(A, Wb, bias, res, Cp, M, N, K, act, outBf16, permTB);
  };

  // ---- feature extractor ----
  k_conv1<<<NB * NT1, 256, 0, stream>>>(feats, conv1_w, conv1_b, c1buf);
  k_conv2<<<NB * NT, 256, 0, stream>>>(c1buf, conv2_w, conv2_b, bufA);
  k_ln<<<BTROWS, 256, 0, stream>>>(bufA, feat_g, feat_b, bufX, 0);
  k_posconv<<<2048, 256, 0, stream>>>(bufX, pos_w, pos_b, bufA);
  k_ln<<<BTROWS, 256, 0, stream>>>(bufA, ctx_g, ctx_b, bufX, 0);

  // ---- transformer layers ----
  for (int i = 0; i < NL; i++) {
    k_ln<<<BTROWS, 256, 0, stream>>>(bufX, ln1g + i * ND, ln1b + i * ND, bufH, 1);
    gemm(bufH, wqkv_bf + (size_t)i * 3 * ND * ND, bqkv + i * 3 * ND,
         nullptr, bufQKV, BTROWS, 3 * ND, ND, 0, 1, 0);
    dim3 ag((NT + 63) / 64, NH, NB);
    k_attn<<<ag, 128, 0, stream>>>(bufQKV, bufO, alibi);
    gemm(bufO, wo_bf + (size_t)i * ND * ND, bo + i * ND,
         bufX, bufX, BTROWS, ND, ND, 0, 0, 0);
    k_ln<<<BTROWS, 256, 0, stream>>>(bufX, ln2g + i * ND, ln2b + i * ND, bufH, 1);
    gemm(bufH, w1_bf + (size_t)i * NFF * ND, b1 + i * NFF,
         nullptr, bufFF, BTROWS, NFF, ND, 1, 1, 0);
    gemm(bufFF, w2_bf + (size_t)i * ND * NFF, b2 + i * ND,
         bufX, bufX, BTROWS, ND, NFF, 0, 0, 0);
  }

  // ---- final projection with [B,T]->[T,B] row permutation ----
  k_f32_to_bf16<<<2048, 256, 0, stream>>>(bufX, bufH, (long)BTROWS * ND);
  gemm(bufH, proj_bf, projb, nullptr, (float*)d_out, BTROWS, NV, ND, 0, 0, 1);
}
